// VisionModel_90202903150645
// MI455X (gfx1250) — compile-verified, matrix operand order checked by reading, not run on hardware
//
#include <hip/hip_runtime.h>
#include <hip/hip_bf16.h>
#include <math.h>

// ---------------------------------------------------------------------------
// Qwen2-VL vision encoder forward on gfx1250 (MI455X).
// bf16 WMMA everywhere; GEMMs use async global->LDS double buffering.
// ---------------------------------------------------------------------------

typedef __bf16 bf16;
typedef __attribute__((ext_vector_type(16))) __bf16 bf16x16;
typedef __attribute__((ext_vector_type(8)))  __bf16 bf16x8;
typedef __attribute__((ext_vector_type(8)))  float  v8f;

#define N_TOK   2048
#define EMBED   1280
#define HEADS   16
#define HD      80
#define HDP     96      // head_dim padded to multiple of 32
#define MLP_H   5120
#define HIDDEN  1536
#define PIN     1176
#define PINP    1184    // patch-in padded to multiple of 32
#define SEG     1024    // block-diagonal attention segment (per image)

// ------------------------------- WMMA helpers ------------------------------

__device__ __forceinline__ v8f wmma_bf16(bf16x16 a, bf16x16 b, v8f c) {
  // D = A(16x32 bf16) * B(32x16 bf16) + C(16x16 f32)
  return __builtin_amdgcn_wmma_f32_16x16x32_bf16(false, a, false, b, (short)0, c,
                                                 false, false);
}

// Load a 16x32 (A) or 32x16 (B) bf16 fragment.
// ISA layout (16-bit, 16x32): lanes 0-15 hold K {0..7,16..23}, lanes 16-31
// hold K {8..15,24..31} of the same 16 rows/cols -> two contiguous b128 loads
// per lane at element offsets (k0 + hi*8) and (k0 + hi*8 + 16).
__device__ __forceinline__ bf16x16 load_frag(const bf16* __restrict__ base,
                                             int ld, int k0, int lane) {
  int r  = lane & 15;
  int hi = (lane >> 4) & 1;
  const bf16* p = base + (size_t)r * ld + k0 + hi * 8;
  bf16x8 lo = *(const bf16x8*)p;
  bf16x8 up = *(const bf16x8*)(p + 16);
  bf16x16 f;
#pragma unroll
  for (int i = 0; i < 8; ++i) { f[i] = lo[i]; f[i + 8] = up[i]; }
  return f;
}

// Async global -> LDS copy, 16 bytes per lane (tracked by ASYNCcnt).
__device__ __forceinline__ void async_ld_b128(unsigned lds_off, const void* gaddr) {
  asm volatile("global_load_async_to_lds_b128 %0, %1, off"
               :: "v"(lds_off), "v"(gaddr) : "memory");
}
__device__ __forceinline__ void wait_async0() {
  asm volatile("s_wait_asynccnt 0" ::: "memory");
}
// Generic->LDS offset: shared-aperture flat addresses carry the LDS byte
// offset in the low 32 bits (aperture selected by addr[63:32]).
__device__ __forceinline__ unsigned lds_off_of(const void* p) {
  return (unsigned)(uintptr_t)p;
}

__device__ __forceinline__ float rmax16(float v) {
  v = fmaxf(v, __shfl_xor(v, 1, 16));
  v = fmaxf(v, __shfl_xor(v, 2, 16));
  v = fmaxf(v, __shfl_xor(v, 4, 16));
  v = fmaxf(v, __shfl_xor(v, 8, 16));
  return v;
}
__device__ __forceinline__ float rsum16(float v) {
  v += __shfl_xor(v, 1, 16);
  v += __shfl_xor(v, 2, 16);
  v += __shfl_xor(v, 4, 16);
  v += __shfl_xor(v, 8, 16);
  return v;
}

__device__ __forceinline__ float gelu_exact(float v) {
  return 0.5f * v * (1.0f + erff(v * 0.70710678118654752f));
}

// ------------------------ f32 -> bf16 (padded) convert ----------------------

__global__ __launch_bounds__(256)
void cvt_pad_kernel(const float* __restrict__ src, bf16* __restrict__ dst,
                    long rows, int K, int Kp) {
  long total = rows * (long)Kp;
  for (long i = blockIdx.x * (long)blockDim.x + threadIdx.x; i < total;
       i += (long)gridDim.x * blockDim.x) {
    long r = i / Kp;
    int  c = (int)(i - r * Kp);
    dst[i] = (c < K) ? (bf16)src[r * (long)K + c] : (bf16)0.0f;
  }
}

// ------------------------------- LayerNorm ---------------------------------

__global__ __launch_bounds__(256)
void ln_kernel(const float* __restrict__ x, const float* __restrict__ w,
               const float* __restrict__ b, bf16* __restrict__ out, int D) {
  __shared__ float rs[256], rs2[256];
  int row = blockIdx.x, tid = threadIdx.x;
  const float* xr = x + (size_t)row * D;
  float s = 0.f, s2 = 0.f;
  for (int i = tid; i < D; i += 256) { float v = xr[i]; s += v; s2 += v * v; }
  rs[tid] = s; rs2[tid] = s2;
  __syncthreads();
  for (int off = 128; off > 0; off >>= 1) {
    if (tid < off) { rs[tid] += rs[tid + off]; rs2[tid] += rs2[tid + off]; }
    __syncthreads();
  }
  float mean = rs[0] / D;
  float var  = rs2[0] / D - mean * mean;
  float inv  = rsqrtf(var + 1e-6f);
  for (int i = tid; i < D; i += 256)
    out[(size_t)row * D + i] = (bf16)((xr[i] - mean) * inv * w[i] + b[i]);
}

// ------------------------------ WMMA GEMM ----------------------------------
// C[M,Nout] = epilogue( A_bf16[M,K] @ W_bf16[Nout,K]^T )
// EPI: 0 none, 1 +bias, 2 +bias+GELU, 3 +bias+residual.  OUTBF: bf16 output.
//
// Block tile 128x128, 8 waves (2 M x 4 N), wave tile 64x32 (8 accum tiles).
// Per K-step (32): A/B 128x32 bf16 slabs staged in LDS via
// global_load_async_to_lds_b128 with double buffering:
//   issue(buf) -> [loop: s_wait_asynccnt 0; s_barrier; issue(buf^1); wmma(buf)]

template <int EPI, bool OUTBF>
__global__ __launch_bounds__(256)
void gemm_wmma(const bf16* __restrict__ A, const bf16* __restrict__ W,
               const float* __restrict__ bias, const float* __restrict__ resid,
               float* __restrict__ Cf, bf16* __restrict__ Cb,
               int M, int Nout, int K) {
  __shared__ bf16 shA[2][128 * 32];
  __shared__ bf16 shB[2][128 * 32];

  const int t    = threadIdx.x;
  const int lane = t & 31;
  const int wave = t >> 5;
  const int hi   = (lane >> 4) & 1;
  const int l16  = lane & 15;
  const int wm   = wave >> 2;          // 0..1 -> 64-row half
  const int wn   = wave & 3;           // 0..3 -> 32-col strip
  const int rowA0 = blockIdx.y * 128;  // block M origin
  const int colB0 = blockIdx.x * 128;  // block N origin

  // copy mapping: 512 chunks of 16B per 8KB slab; 2 chunks/thread
  const int c0   = t * 2;
  const int cr0  = c0 >> 2,      cs0 = (c0 & 3) * 8;   // row, elem offset
  const int cr1  = (c0 + 1) >> 2, cs1 = ((c0 + 1) & 3) * 8;

  auto issue = [&](int buf, int k0) {
    async_ld_b128(lds_off_of(&shA[buf][cr0 * 32 + cs0]),
                  A + (size_t)(rowA0 + cr0) * K + k0 + cs0);
    async_ld_b128(lds_off_of(&shB[buf][cr0 * 32 + cs0]),
                  W + (size_t)(colB0 + cr0) * K + k0 + cs0);
    async_ld_b128(lds_off_of(&shA[buf][cr1 * 32 + cs1]),
                  A + (size_t)(rowA0 + cr1) * K + k0 + cs1);
    async_ld_b128(lds_off_of(&shB[buf][cr1 * 32 + cs1]),
                  W + (size_t)(colB0 + cr1) * K + k0 + cs1);
  };

  v8f acc[4][2] = {};
  const int nk = K >> 5;
  issue(0, 0);
  int buf = 0;
  for (int ks = 0; ks < nk; ++ks) {
    wait_async0();          // this wave's copies into shA/shB[buf] landed
    __syncthreads();        // ... and everyone else's
    if (ks + 1 < nk) issue(buf ^ 1, (ks + 1) << 5);  // overlap next slab

    bf16x16 af[4];
#pragma unroll
    for (int mi = 0; mi < 4; ++mi)
      af[mi] = load_frag(&shA[buf][(wm * 64 + mi * 16) * 32], 32, 0, lane);
#pragma unroll
    for (int ni = 0; ni < 2; ++ni) {
      bf16x16 bfr = load_frag(&shB[buf][(wn * 32 + ni * 16) * 32], 32, 0, lane);
#pragma unroll
      for (int mi = 0; mi < 4; ++mi)
        acc[mi][ni] = wmma_bf16(af[mi], bfr, acc[mi][ni]);
    }
    buf ^= 1;
  }

#pragma unroll
  for (int mi = 0; mi < 4; ++mi) {
#pragma unroll
    for (int ni = 0; ni < 2; ++ni) {
#pragma unroll
      for (int r = 0; r < 8; ++r) {
        int m = rowA0 + wm * 64 + mi * 16 + r + hi * 8;
        int n = colB0 + wn * 32 + ni * 16 + l16;
        float v = acc[mi][ni][r];
        if (EPI >= 1) v += bias[n];
        if (EPI == 2) v = gelu_exact(v);
        if (EPI == 3) v += resid[(size_t)m * Nout + n];
        if (OUTBF) Cb[(size_t)m * Nout + n] = (bf16)v;
        else       Cf[(size_t)m * Nout + n] = v;
      }
    }
  }
}

// ------------------- RoPE + qkv split + pad + V transpose -------------------
// qkv f32 [N, 3, HEADS, HD] -> Qb,Kb bf16 [H][N][HDP], Vt bf16 [H][HDP][N].
// cos/sin recomputed analytically from the static 32x32 merge-2 grid.

__global__ __launch_bounds__(256)
void rope_split_kernel(const float* __restrict__ qkv, bf16* __restrict__ Qb,
                       bf16* __restrict__ Kb, bf16* __restrict__ Vt, int Nn) {
  int idx = blockIdx.x * 256 + threadIdx.x;
  int total = Nn * HEADS * HDP;
  if (idx >= total) return;
  int d = idx % HDP;
  int h = (idx / HDP) % HEADS;
  int n = idx / (HDP * HEADS);

  bf16 qo, ko, vo;
  if (d < HD) {
    int i  = n % SEG;                        // position inside 32x32 image
    int wi = i & 1, hh = (i >> 1) & 1, wb = (i >> 2) & 15, hb = (i >> 6) & 15;
    float hpos = (float)(hb * 2 + hh), wpos = (float)(wb * 2 + wi);
    int   k40  = d % 40;                     // rotary dim = HD/2 = 40, tiled x2
    int   j    = k40 % 20;
    float invf = __powf(10000.0f, -((float)(2 * j)) / 40.0f);
    float fr   = ((k40 < 20) ? hpos : wpos) * invf;
    float c = __cosf(fr), s = __sinf(fr);
    int   d2  = (d < 40) ? d + 40 : d - 40;
    float sgn = (d < 40) ? -1.f : 1.f;
    size_t rowq = (size_t)n * (3 * EMBED);
    float q  = qkv[rowq + 0 * EMBED + h * HD + d];
    float qr = sgn * qkv[rowq + 0 * EMBED + h * HD + d2];
    float k  = qkv[rowq + 1 * EMBED + h * HD + d];
    float kr = sgn * qkv[rowq + 1 * EMBED + h * HD + d2];
    float v  = qkv[rowq + 2 * EMBED + h * HD + d];
    qo = (bf16)(q * c + qr * s);
    ko = (bf16)(k * c + kr * s);
    vo = (bf16)v;
  } else {
    qo = (bf16)0.0f; ko = (bf16)0.0f; vo = (bf16)0.0f;
  }
  Qb[(size_t)h * Nn * HDP + (size_t)n * HDP + d] = qo;
  Kb[(size_t)h * Nn * HDP + (size_t)n * HDP + d] = ko;
  Vt[(size_t)h * HDP * Nn + (size_t)d * Nn + n] = vo;
}

// --------------------------- Flash attention -------------------------------
// Block-diagonal mask == independent SEG=1024 windows. Grid (N/64, HEADS),
// 4 waves/block, each wave owns 16 q-rows, online softmax, WMMA throughout.

__global__ __launch_bounds__(128)
void attn_kernel(const bf16* __restrict__ Qb, const bf16* __restrict__ Kb,
                 const bf16* __restrict__ Vt, bf16* __restrict__ outb, int Nn) {
  __shared__ bf16 shp[4 * 16 * 32];          // per-wave P tile (C->A relayout)
  int lane = threadIdx.x & 31, wave = threadIdx.x >> 5;
  int hi = lane >> 4, l16 = lane & 15;
  int h  = blockIdx.y;
  int q0 = blockIdx.x * 64 + wave * 16;
  int kb0 = (q0 / SEG) * SEG;

  const bf16* Qh = Qb + (size_t)h * Nn * HDP;
  const bf16* Kh = Kb + (size_t)h * Nn * HDP;
  const bf16* Vh = Vt + (size_t)h * HDP * Nn;

  bf16x16 qf0 = load_frag(Qh + (size_t)q0 * HDP, HDP, 0, lane);
  bf16x16 qf1 = load_frag(Qh + (size_t)q0 * HDP, HDP, 32, lane);
  bf16x16 qf2 = load_frag(Qh + (size_t)q0 * HDP, HDP, 64, lane);

  v8f o[6] = {};
  float mrow[8], lrow[8];
#pragma unroll
  for (int r = 0; r < 8; ++r) { mrow[r] = -3.0e38f; lrow[r] = 0.f; }
  const float scale = 0.11180339887498948f;  // 1/sqrt(80)
  bf16* myp = shp + wave * 16 * 32;

  for (int kk = 0; kk < SEG; kk += 32) {
    const bf16* Kt0 = Kh + (size_t)(kb0 + kk) * HDP;
    const bf16* Kt1 = Kt0 + 16 * HDP;
    v8f s0 = {}, s1 = {};
    s0 = wmma_bf16(qf0, load_frag(Kt0, HDP, 0,  lane), s0);
    s0 = wmma_bf16(qf1, load_frag(Kt0, HDP, 32, lane), s0);
    s0 = wmma_bf16(qf2, load_frag(Kt0, HDP, 64, lane), s0);
    s1 = wmma_bf16(qf0, load_frag(Kt1, HDP, 0,  lane), s1);
    s1 = wmma_bf16(qf1, load_frag(Kt1, HDP, 32, lane), s1);
    s1 = wmma_bf16(qf2, load_frag(Kt1, HDP, 64, lane), s1);

    float p0[8], p1[8], alpha[8];
#pragma unroll
    for (int r = 0; r < 8; ++r) {
      float a = s0[r] * scale, b = s1[r] * scale;
      float mx = rmax16(fmaxf(a, b));
      mx = fmaxf(mx, mrow[r]);
      alpha[r] = __expf(mrow[r] - mx);
      mrow[r] = mx;
      p0[r] = __expf(a - mx);
      p1[r] = __expf(b - mx);
      lrow[r] = lrow[r] * alpha[r] + p0[r] + p1[r];  // per-lane partial
    }
#pragma unroll
    for (int tt = 0; tt < 6; ++tt)
#pragma unroll
      for (int r = 0; r < 8; ++r) o[tt][r] *= alpha[r];

    // C-layout -> A-layout via wave-private LDS tile
#pragma unroll
    for (int r = 0; r < 8; ++r) {
      int m = r + hi * 8;
      myp[m * 32 + l16]      = (bf16)p0[r];
      myp[m * 32 + 16 + l16] = (bf16)p1[r];
    }
    asm volatile("s_wait_dscnt 0" ::: "memory");
    bf16x16 pf = load_frag(myp, 32, 0, lane);

#pragma unroll
    for (int tt = 0; tt < 6; ++tt) {
      bf16x16 vf = load_frag(Vh + (size_t)tt * 16 * Nn + kb0 + kk, Nn, 0, lane);
      o[tt] = wmma_bf16(pf, vf, o[tt]);
    }
  }

#pragma unroll
  for (int r = 0; r < 8; ++r) lrow[r] = rsum16(lrow[r]);
#pragma unroll
  for (int r = 0; r < 8; ++r) {
    float inv = 1.0f / lrow[r];
    int m = q0 + r + hi * 8;
#pragma unroll
    for (int tt = 0; tt < 5; ++tt)  // drop padded cols 80..95
      outb[(size_t)m * EMBED + h * HD + tt * 16 + l16] = (bf16)(o[tt][r] * inv);
  }
}

// ------------------------------- host side ---------------------------------

static inline int cdiv_i(long a, long b) { return (int)((a + b - 1) / b); }

extern "C" void kernel_launch(void* const* d_in, const int* in_sizes, int n_in,
                              void* d_out, int out_size, void* d_ws, size_t ws_size,
                              hipStream_t stream) {
  (void)in_sizes; (void)n_in; (void)out_size; (void)ws_size;

  // Inputs in setup_inputs() dict order:
  // 0 hidden_states, 1 grid_thw, 2 patch_w,
  // then 4 blocks x {ln1_w,ln1_b,ln2_w,ln2_b,qkv_w,qkv_b,proj_w,proj_b,
  //                  fc1_w,fc1_b,fc2_w,fc2_b},
  // then lnq_w, lnq_b, m1_w, m1_b, m2_w, m2_b.
  const float* hs      = (const float*)d_in[0];
  const float* patch_w = (const float*)d_in[2];
  const int mb = 3 + 4 * 12;
  const float* lnq_w = (const float*)d_in[mb + 0];
  const float* lnq_b = (const float*)d_in[mb + 1];
  const float* m1_w  = (const float*)d_in[mb + 2];
  const float* m1_b  = (const float*)d_in[mb + 3];
  const float* m2_w  = (const float*)d_in[mb + 4];
  const float* m2_b  = (const float*)d_in[mb + 5];

  // Workspace bump allocator (256B aligned).
  char* wp = (char*)d_ws;
  auto alloc = [&](size_t bytes) -> void* {
    void* r = (void*)wp;
    wp += (bytes + 255) & ~(size_t)255;
    return r;
  };

  bf16*  hs_b    = (bf16*)alloc((size_t)N_TOK * PINP * 2);
  bf16*  patchwb = (bf16*)alloc((size_t)EMBED * PINP * 2);
  float* x       = (float*)alloc((size_t)N_TOK * EMBED * 4);
  bf16*  lnbuf   = (bf16*)alloc((size_t)N_TOK * EMBED * 2);
  float* qkvbuf  = (float*)alloc((size_t)N_TOK * 3 * EMBED * 4);
  bf16*  Qb      = (bf16*)alloc((size_t)HEADS * N_TOK * HDP * 2);
  bf16*  Kb      = (bf16*)alloc((size_t)HEADS * N_TOK * HDP * 2);
  bf16*  Vt      = (bf16*)alloc((size_t)HEADS * HDP * N_TOK * 2);
  bf16*  attnout = (bf16*)alloc((size_t)N_TOK * EMBED * 2);
  bf16*  h1      = (bf16*)alloc((size_t)N_TOK * MLP_H * 2);
  // Per-block weight arena (re-converted each layer; deterministic).
  bf16*  wqkv    = (bf16*)alloc((size_t)3 * EMBED * EMBED * 2);
  bf16*  wproj   = (bf16*)alloc((size_t)EMBED * EMBED * 2);
  bf16*  wfc1    = (bf16*)alloc((size_t)MLP_H * EMBED * 2);
  bf16*  wfc2    = (bf16*)alloc((size_t)EMBED * MLP_H * 2);
  bf16*  m1b     = (bf16*)alloc((size_t)MLP_H * MLP_H * 2);
  bf16*  m2b     = (bf16*)alloc((size_t)HIDDEN * MLP_H * 2);
  bf16*  mrg     = (bf16*)alloc((size_t)(N_TOK / 4) * MLP_H * 2);

  auto cvt = [&](const float* s, bf16* d, long rows, int K, int Kp) {
    long total = rows * (long)Kp;
    cvt_pad_kernel<<<cdiv_i(total, 256), 256, 0, stream>>>(s, d, rows, K, Kp);
  };

  // Patch embed: x = hs @ patch_w^T  [2048,1280]
  cvt(hs, hs_b, N_TOK, PIN, PINP);
  cvt(patch_w, patchwb, EMBED, PIN, PINP);
  gemm_wmma<0, false><<<dim3(EMBED / 128, N_TOK / 128), 256, 0, stream>>>(
      hs_b, patchwb, nullptr, nullptr, x, nullptr, N_TOK, EMBED, PINP);

  for (int blk = 0; blk < 4; ++blk) {
    const int base = 3 + blk * 12;
    const float* ln1_w  = (const float*)d_in[base + 0];
    const float* ln1_b  = (const float*)d_in[base + 1];
    const float* ln2_w  = (const float*)d_in[base + 2];
    const float* ln2_b  = (const float*)d_in[base + 3];
    const float* qkv_w  = (const float*)d_in[base + 4];
    const float* qkv_b  = (const float*)d_in[base + 5];
    const float* proj_w = (const float*)d_in[base + 6];
    const float* proj_b = (const float*)d_in[base + 7];
    const float* fc1_w  = (const float*)d_in[base + 8];
    const float* fc1_b  = (const float*)d_in[base + 9];
    const float* fc2_w  = (const float*)d_in[base + 10];
    const float* fc2_b  = (const float*)d_in[base + 11];

    cvt(qkv_w,  wqkv,  3 * EMBED, EMBED, EMBED);
    cvt(proj_w, wproj, EMBED,     EMBED, EMBED);
    cvt(fc1_w,  wfc1,  MLP_H,     EMBED, EMBED);
    cvt(fc2_w,  wfc2,  EMBED,     MLP_H, MLP_H);

    // y = LN1(x); qkv = y @ qkv_w^T + qkv_b
    ln_kernel<<<N_TOK, 256, 0, stream>>>(x, ln1_w, ln1_b, lnbuf, EMBED);
    gemm_wmma<1, false><<<dim3(3 * EMBED / 128, N_TOK / 128), 256, 0, stream>>>(
        lnbuf, wqkv, qkv_b, nullptr, qkvbuf, nullptr, N_TOK, 3 * EMBED, EMBED);

    rope_split_kernel<<<cdiv_i((long)N_TOK * HEADS * HDP, 256), 256, 0, stream>>>(
        qkvbuf, Qb, Kb, Vt, N_TOK);
    attn_kernel<<<dim3(N_TOK / 64, HEADS), 128, 0, stream>>>(Qb, Kb, Vt, attnout, N_TOK);

    // x = x + attn @ proj_w^T + proj_b
    gemm_wmma<3, false><<<dim3(EMBED / 128, N_TOK / 128), 256, 0, stream>>>(
        attnout, wproj, proj_b, x, x, nullptr, N_TOK, EMBED, EMBED);

    // MLP: x = x + fc2(gelu(fc1(LN2(x))))
    ln_kernel<<<N_TOK, 256, 0, stream>>>(x, ln2_w, ln2_b, lnbuf, EMBED);
    gemm_wmma<2, true><<<dim3(MLP_H / 128, N_TOK / 128), 256, 0, stream>>>(
        lnbuf, wfc1, fc1_b, nullptr, nullptr, h1, N_TOK, MLP_H, EMBED);
    gemm_wmma<3, false><<<dim3(EMBED / 128, N_TOK / 128), 256, 0, stream>>>(
        h1, wfc2, fc2_b, x, x, nullptr, N_TOK, EMBED, MLP_H);
  }

  // PatchMerger: LN(x) reshaped [512, 5120] -> gelu(m1) -> m2 -> d_out
  ln_kernel<<<N_TOK, 256, 0, stream>>>(x, lnq_w, lnq_b, lnbuf, EMBED);
  cvt(m1_w, m1b, MLP_H, MLP_H, MLP_H);
  cvt(m2_w, m2b, HIDDEN, MLP_H, MLP_H);
  gemm_wmma<2, true><<<dim3(MLP_H / 128, (N_TOK / 4) / 128), 256, 0, stream>>>(
      lnbuf /* contiguous reshape [512,5120] */, m1b, m1_b, nullptr, nullptr,
      mrg, N_TOK / 4, MLP_H, MLP_H);
  gemm_wmma<1, false><<<dim3(HIDDEN / 128, (N_TOK / 4) / 128), 256, 0, stream>>>(
      mrg, m2b, m2_b, nullptr, (float*)d_out, nullptr, N_TOK / 4, HIDDEN, MLP_H);
}
